// Attention_62861141344534
// MI455X (gfx1250) — matrix operand order repaired and checked by reading, not compiled, not run on hardware
//
#include <hip/hip_runtime.h>
#include <cstdint>
#include <cstddef>

// ---------------------------------------------------------------------------
// Types for CDNA5 WMMA (wave32, V_WMMA_F32_16X16X32_BF16)
// ---------------------------------------------------------------------------
typedef __bf16 v16bf __attribute__((ext_vector_type(16)));
typedef __bf16 v8bf  __attribute__((ext_vector_type(8)));
typedef float  v8f   __attribute__((ext_vector_type(8)));

#define Bc  2
#define Sc  2048
#define Dc  2048
#define Hc  16
#define KVc 8
#define HDc 128

// Use the CDNA5 Tensor Data Mover when the toolchain exposes it.
#if defined(__HIP_DEVICE_COMPILE__) && \
    __has_builtin(__builtin_amdgcn_tensor_load_to_lds) && \
    __has_builtin(__builtin_amdgcn_s_wait_tensorcnt)
#define USE_TDM 1
#else
#define USE_TDM 0
#endif

#if USE_TDM
typedef unsigned int tdm_v4u __attribute__((ext_vector_type(4)));
typedef int          tdm_v4i __attribute__((ext_vector_type(4)));
typedef int          tdm_v8i __attribute__((ext_vector_type(8)));

// Low 32 bits of a generic pointer to a __shared__ object = LDS byte offset.
__device__ __forceinline__ unsigned lds_off32(const void* p) {
  return (unsigned)(unsigned long long)p;
}

// 2-D TDM tile load: tile_d1 rows x tile_d0 bf16 elements, global row stride
// stride_elems, LDS destination with hardware row padding. TENSORcnt-tracked;
// EXEC-independent; issue from one wave only.
__device__ __forceinline__ void tdm_load_2d(const void* gbase,
                                            unsigned lds_byte_off,
                                            unsigned tile_d0, unsigned tile_d1,
                                            unsigned stride_elems,
                                            unsigned pad_interval,
                                            unsigned pad_amount) {
  unsigned long long ga = (unsigned long long)gbase;
  tdm_v4u g0;
  g0.x = 1u;                                        // count=1, user mode
  g0.y = lds_byte_off;                              // lds_addr
  g0.z = (unsigned)(ga & 0xffffffffull);            // global_addr[31:0]
  g0.w = (unsigned)((ga >> 32) & 0x01ffffffull)     // global_addr[56:32]
       | 0x80000000u;                               // type=2 ("image")

  unsigned td0 = stride_elems;                      // in-bounds by design
  unsigned td1 = 1u << 30;                          // OOB effectively off
  tdm_v8i g1;
  g1[0] = (int)((1u << 16)                          // data_size = 2 bytes
              | (1u << 20)                          // pad_enable
              | (pad_interval << 22)
              | (pad_amount << 25));
  g1[1] = (int)((td0 & 0xffffu) << 16);
  g1[2] = (int)(((td0 >> 16) & 0xffffu) | ((td1 & 0xffffu) << 16));
  g1[3] = (int)(((td1 >> 16) & 0xffffu) | ((tile_d0 & 0xffffu) << 16));
  g1[4] = (int)(tile_d1 & 0xffffu);
  g1[5] = (int)stride_elems;
  g1[6] = 0;
  g1[7] = 0;
  tdm_v4i z4 = {0, 0, 0, 0};
#if __clang_major__ >= 23
  tdm_v8i z8 = {0, 0, 0, 0, 0, 0, 0, 0};
  __builtin_amdgcn_tensor_load_to_lds(g0, g1, z4, z4, z8, 0);
#else
  __builtin_amdgcn_tensor_load_to_lds(g0, g1, z4, z4, 0);
#endif
}
#endif  // USE_TDM

__device__ __forceinline__ v8f wmma_bf16(v16bf a, v16bf b, v8f c) {
  return __builtin_amdgcn_wmma_f32_16x16x32_bf16(
      false, a, false, b, (short)0, c, false, false);
}

// A-fragment (16x32, row-major source with leading dim ld, bf16).
__device__ __forceinline__ v16bf load_a_frag(const __bf16* base, int ld) {
  int lane = threadIdx.x & 31;
  int row  = lane & 15;
  int kb   = (lane >> 4) * 16;
  const __bf16* p = base + (size_t)row * ld;
  v16bf a;
#pragma unroll
  for (int i = 0; i < 4; ++i) {
    a[2 * i]         = p[kb + 2 * i];
    a[2 * i + 1]     = p[kb + 2 * i + 1];
    a[8 + 2 * i]     = p[8 + kb + 2 * i];
    a[8 + 2 * i + 1] = p[8 + kb + 2 * i + 1];
  }
  return a;
}

// B-fragment for logical B(k,n) from a transposed (n-major) source.
__device__ __forceinline__ v16bf load_bT_frag(const __bf16* base, int ld) {
  int lane = threadIdx.x & 31;
  int n    = lane & 15;
  int kb   = (lane >> 4) * 16;
  const __bf16* p = base + (size_t)n * ld + kb;
  v16bf b;
#pragma unroll
  for (int j = 0; j < 8; ++j) {
    b[2 * j]     = p[2 * j];
    b[2 * j + 1] = p[2 * j + 1];
  }
  return b;
}

// ---------------------------------------------------------------------------
// fp32 -> bf16 (plain)
// ---------------------------------------------------------------------------
__global__ void cvt_f32_bf16(const float* __restrict__ in,
                             __bf16* __restrict__ out, long n) {
  long i = (long)blockIdx.x * blockDim.x + threadIdx.x;
  if (i < n) out[i] = (__bf16)in[i];
}

// ---------------------------------------------------------------------------
// fp32 (R x C) -> bf16 transposed (C x R); tiled through LDS, coalesced.
// ---------------------------------------------------------------------------
__global__ __launch_bounds__(256) void transpose_cvt_f32_bf16(
    const float* __restrict__ in, __bf16* __restrict__ out, int R, int C) {
  __shared__ float tile[32][33];
  const int tx = threadIdx.x & 31;
  const int ty = threadIdx.x >> 5;
  const int r0 = blockIdx.y * 32;
  const int c0 = blockIdx.x * 32;
#pragma unroll
  for (int i = 0; i < 4; ++i)
    tile[ty + 8 * i][tx] = in[(size_t)(r0 + ty + 8 * i) * C + c0 + tx];
  __syncthreads();
#pragma unroll
  for (int i = 0; i < 4; ++i)
    out[(size_t)(c0 + ty + 8 * i) * R + r0 + tx] =
        (__bf16)tile[tx][ty + 8 * i];
}

// ---------------------------------------------------------------------------
// bf16 transpose of V: (b, s, n=1024) -> Vt (b, n, s); tiled, coalesced.
// ---------------------------------------------------------------------------
__global__ __launch_bounds__(256) void transpose_v_bf16(
    const __bf16* __restrict__ in, __bf16* __restrict__ out) {
  __shared__ __bf16 tile[32][40];
  const int tx = threadIdx.x & 31;
  const int ty = threadIdx.x >> 5;
  const int b  = blockIdx.z;
  const int s0 = blockIdx.y * 32;
  const int n0 = blockIdx.x * 32;
#pragma unroll
  for (int i = 0; i < 4; ++i)
    tile[ty + 8 * i][tx] =
        in[((size_t)b * Sc + s0 + ty + 8 * i) * (KVc * HDc) + n0 + tx];
  __syncthreads();
#pragma unroll
  for (int i = 0; i < 4; ++i)
    out[((size_t)b * (KVc * HDc) + n0 + ty + 8 * i) * Sc + s0 + tx] =
        tile[tx][ty + 8 * i];
}

// ---------------------------------------------------------------------------
// Tiled bf16 GEMM: C(MxN) = A(MxK) * Bt^T, Bt given TRANSPOSED (N x K).
// Compile-time N/K/output-type so the epilogue is branch-free and address
// math strength-reduces. Block tile 128x128, K-step 32, 8 waves, wave tile
// 32x64. TDM path: double-buffered LDS, wave 0 issues the DMA tile loads.
// ---------------------------------------------------------------------------
template <int N, int K, bool OUTBF16>
__global__ __launch_bounds__(256) void gemm_bf16(
    const __bf16* __restrict__ A, const __bf16* __restrict__ Bt,
    void* __restrict__ Cout) {
  __shared__ __bf16 sA [2][128][40];   // 32 k + 8 pad (TDM row pad 16B)
  __shared__ __bf16 sBt[2][128][40];

  const int t    = threadIdx.x;
  const int wave = t >> 5;
  const int lane = t & 31;
  const int m0   = blockIdx.y * 128;
  const int n0   = blockIdx.x * 128;
  const int wm   = (wave >> 1) * 32;
  const int wn   = (wave & 1) * 64;

  v8f acc[2][4];
#pragma unroll
  for (int i = 0; i < 2; ++i)
#pragma unroll
    for (int j = 0; j < 4; ++j) acc[i][j] = (v8f)0.0f;

#if USE_TDM
  if (wave == 0) {
    tdm_load_2d(A  + (size_t)m0 * K, lds_off32(&sA [0][0][0]), 32, 128, K, 3, 3);
    tdm_load_2d(Bt + (size_t)n0 * K, lds_off32(&sBt[0][0][0]), 32, 128, K, 3, 3);
    __builtin_amdgcn_s_wait_tensorcnt(0);
  }
  __syncthreads();

  for (int k0 = 0; k0 < K; k0 += 32) {
    const int cur = (k0 >> 5) & 1;
    if (wave == 0 && k0 + 32 < K) {
      tdm_load_2d(A  + (size_t)m0 * K + k0 + 32,
                  lds_off32(&sA [cur ^ 1][0][0]), 32, 128, K, 3, 3);
      tdm_load_2d(Bt + (size_t)n0 * K + k0 + 32,
                  lds_off32(&sBt[cur ^ 1][0][0]), 32, 128, K, 3, 3);
    }
    v16bf af[2];
#pragma unroll
    for (int i = 0; i < 2; ++i)
      af[i] = load_a_frag(&sA[cur][wm + 16 * i][0], 40);
    v16bf bfr[4];
#pragma unroll
    for (int j = 0; j < 4; ++j)
      bfr[j] = load_bT_frag(&sBt[cur][wn + 16 * j][0], 40);
#pragma unroll
    for (int i = 0; i < 2; ++i)
#pragma unroll
      for (int j = 0; j < 4; ++j)
        acc[i][j] = wmma_bf16(af[i], bfr[j], acc[i][j]);

    if (wave == 0) __builtin_amdgcn_s_wait_tensorcnt(0);
    __syncthreads();
  }
#else
  v8bf ar[2], br[2];
#pragma unroll
  for (int c = 0; c < 2; ++c) {
    int idx = c * 2048 + t * 8;
    int r = idx >> 5, col = idx & 31;
    ar[c] = *(const v8bf*)&A [(size_t)(m0 + r) * K + col];
    br[c] = *(const v8bf*)&Bt[(size_t)(n0 + r) * K + col];
  }
  for (int k0 = 0; k0 < K; k0 += 32) {
    __syncthreads();
#pragma unroll
    for (int c = 0; c < 2; ++c) {
      int idx = c * 2048 + t * 8;
      int r = idx >> 5, col = idx & 31;
      *(v8bf*)&sA [0][r][col] = ar[c];
      *(v8bf*)&sBt[0][r][col] = br[c];
    }
    __syncthreads();
    if (k0 + 32 < K) {
#pragma unroll
      for (int c = 0; c < 2; ++c) {
        int idx = c * 2048 + t * 8;
        int r = idx >> 5, col = idx & 31;
        ar[c] = *(const v8bf*)&A [(size_t)(m0 + r) * K + k0 + 32 + col];
        br[c] = *(const v8bf*)&Bt[(size_t)(n0 + r) * K + k0 + 32 + col];
      }
    }
    v16bf af[2];
#pragma unroll
    for (int i = 0; i < 2; ++i)
      af[i] = load_a_frag(&sA[0][wm + 16 * i][0], 40);
    v16bf bfr[4];
#pragma unroll
    for (int j = 0; j < 4; ++j)
      bfr[j] = load_bT_frag(&sBt[0][wn + 16 * j][0], 40);
#pragma unroll
    for (int i = 0; i < 2; ++i)
#pragma unroll
      for (int j = 0; j < 4; ++j)
        acc[i][j] = wmma_bf16(af[i], bfr[j], acc[i][j]);
  }
#endif

  // epilogue: lane owns col N = lane&15, rows r + 8*(lane>>4); branch-free,
  // base pointer + constant-stride stores per 16x16 subtile.
  const int cn    = lane & 15;
  const int rbase = (lane >> 4) * 8;
#pragma unroll
  for (int i = 0; i < 2; ++i)
#pragma unroll
    for (int j = 0; j < 4; ++j) {
      const size_t row0 = (size_t)(m0 + wm + 16 * i + rbase);
      const size_t col  = (size_t)(n0 + wn + 16 * j + cn);
      if (OUTBF16) {
        __bf16* p = (__bf16*)Cout + row0 * N + col;
#pragma unroll
        for (int r = 0; r < 8; ++r) p[(size_t)r * N] = (__bf16)acc[i][j][r];
      } else {
        float* p = (float*)Cout + row0 * N + col;
#pragma unroll
        for (int r = 0; r < 8; ++r) p[(size_t)r * N] = acc[i][j][r];
      }
    }
}

// ---------------------------------------------------------------------------
// RoPE, in place on bf16 (B, S, heads, 128).
// ---------------------------------------------------------------------------
__global__ void rope_kernel(__bf16* __restrict__ X,
                            const float* __restrict__ cosb,
                            const float* __restrict__ sinb, int heads) {
  long i = (long)blockIdx.x * blockDim.x + threadIdx.x;
  long total = (long)Bc * Sc * heads * 64;
  if (i >= total) return;
  int  d    = (int)(i & 63);
  long rest = i >> 6;
  int  hh   = (int)(rest % heads);
  long bs   = rest / heads;
  int  s    = (int)(bs % Sc);
  size_t off = ((size_t)bs * heads + hh) * HDc;
  float x1 = (float)X[off + d];
  float x2 = (float)X[off + d + 64];
  float c1 = cosb[(size_t)s * HDc + d];
  float s1 = sinb[(size_t)s * HDc + d];
  float c2 = cosb[(size_t)s * HDc + d + 64];
  float s2 = sinb[(size_t)s * HDc + d + 64];
  X[off + d]      = (__bf16)(x1 * c1 - x2 * s1);
  X[off + d + 64] = (__bf16)(x2 * c2 + x1 * s2);
}

// ---------------------------------------------------------------------------
// Causal GQA flash attention. grid = (S/64, B*H), block = 128 (4 waves);
// each wave owns 16 q rows. K tile (32 keys x 128 hd) and Vt tile
// (128 hd x 32 keys) staged via TDM double-buffering (fallback: reg pipeline).
// ---------------------------------------------------------------------------
__global__ __launch_bounds__(128) void attn_kernel(
    const __bf16* __restrict__ Q, const __bf16* __restrict__ Kc,
    const __bf16* __restrict__ Vt, __bf16* __restrict__ Oc) {
  __shared__ __bf16 sK [2][32][HDc + 8];  // row 256B + 16B pad
  __shared__ __bf16 sVt[2][HDc][40];      // row 64B + 16B pad
  __shared__ float  sS[4][16][33];
  __shared__ __bf16 sP[4][16][40];
  __shared__ float  sM[4][16], sL[4][16], sC[4][16];

  const int qtile = blockIdx.x;
  const int bh    = blockIdx.y;
  const int b     = bh / Hc;
  const int h     = bh % Hc;
  const int kvh   = h / (Hc / KVc);
  const int t     = threadIdx.x;
  const int wave  = t >> 5;
  const int lane  = t & 31;
  const int q0    = qtile * 64 + wave * 16;

  if (lane < 16) { sM[wave][lane] = -1e30f; sL[wave][lane] = 0.0f; }

  const __bf16* Qbase = Q + ((size_t)b * Sc + q0) * (Hc * HDc) + h * HDc;
  v16bf qf[4];
#pragma unroll
  for (int c = 0; c < 4; ++c) qf[c] = load_a_frag(Qbase + c * 32, Hc * HDc);

  v8f o[8];
#pragma unroll
  for (int nt = 0; nt < 8; ++nt) o[nt] = (v8f)0.0f;

  const float scale = 0.08838834764831845f;  // 1/sqrt(128)
  const int   hi    = (lane >> 4) * 8;
  const int   kend  = qtile * 64 + 63;

  const __bf16* Kbase  = Kc + ((size_t)b * Sc) * (KVc * HDc) + kvh * HDc;
  const __bf16* Vtbase = Vt + ((size_t)b * (KVc * HDc) + kvh * HDc) * Sc;

#if USE_TDM
  if (wave == 0) {
    tdm_load_2d(Kbase,  lds_off32(&sK [0][0][0]), 128, 32, KVc * HDc, 5, 3);
    tdm_load_2d(Vtbase, lds_off32(&sVt[0][0][0]), 32, 128, Sc,        3, 3);
    __builtin_amdgcn_s_wait_tensorcnt(0);
  }
  __syncthreads();
#else
  v8bf kreg[4], vreg[4];
  auto fetch = [&](int kk) {
#pragma unroll
    for (int c = 0; c < 4; ++c) {
      int idx = c * 1024 + t * 8;
      int rk = idx >> 7, ck = idx & 127;
      kreg[c] = *(const v8bf*)&Kbase[(size_t)(kk + rk) * (KVc * HDc) + ck];
      int rv = idx >> 5, cv = idx & 31;
      vreg[c] = *(const v8bf*)&Vtbase[(size_t)rv * Sc + kk + cv];
    }
  };
  fetch(0);
#endif

  for (int k0 = 0; k0 <= kend; k0 += 32) {
#if USE_TDM
    const int cur = (k0 >> 5) & 1;
    if (wave == 0 && k0 + 32 <= kend) {
      tdm_load_2d(Kbase + (size_t)(k0 + 32) * (KVc * HDc),
                  lds_off32(&sK [cur ^ 1][0][0]), 128, 32, KVc * HDc, 5, 3);
      tdm_load_2d(Vtbase + k0 + 32,
                  lds_off32(&sVt[cur ^ 1][0][0]), 32, 128, Sc, 3, 3);
    }
    const __bf16 (*pK)[HDc + 8] = sK[cur];
    const __bf16 (*pV)[40]      = sVt[cur];
#else
    __syncthreads();
#pragma unroll
    for (int c = 0; c < 4; ++c) {
      int idx = c * 1024 + t * 8;
      int rk = idx >> 7, ck = idx & 127;
      *(v8bf*)&sK[0][rk][ck] = kreg[c];
      int rv = idx >> 5, cv = idx & 31;
      *(v8bf*)&sVt[0][rv][cv] = vreg[c];
    }
    __syncthreads();
    if (k0 + 32 <= kend) fetch(k0 + 32);
    const __bf16 (*pK)[HDc + 8] = sK[0];
    const __bf16 (*pV)[40]      = sVt[0];
#endif

    // scores S = Q * K^T : 16 x 32 (two 16x16 n-tiles), accumulate over HD
    v8f sc[2];
    sc[0] = (v8f)0.0f; sc[1] = (v8f)0.0f;
#pragma unroll
    for (int c = 0; c < 4; ++c)
#pragma unroll
      for (int nt = 0; nt < 2; ++nt) {
        v16bf kb = load_bT_frag(&pK[nt * 16][c * 32], HDc + 8);
        sc[nt] = wmma_bf16(qf[c], kb, sc[nt]);
      }
#pragma unroll
    for (int nt = 0; nt < 2; ++nt)
#pragma unroll
      for (int r = 0; r < 8; ++r)
        sS[wave][hi + r][nt * 16 + (lane & 15)] = sc[nt][r] * scale;

    // online softmax: lanes 0-15 each own one full row of 32 keys
    if (lane < 16) {
      int row = lane, qg = q0 + row;
      float mprev = sM[wave][row];
      float mnew  = mprev;
#pragma unroll
      for (int c = 0; c < 32; ++c) {
        float v = sS[wave][row][c];
        if ((k0 + c) <= qg && v > mnew) mnew = v;
      }
      float corr = __expf(mprev - mnew);
      float sum  = 0.0f;
#pragma unroll
      for (int c = 0; c < 32; ++c) {
        float e = ((k0 + c) <= qg) ? __expf(sS[wave][row][c] - mnew) : 0.0f;
        sP[wave][row][c] = (__bf16)e;
        sum += e;
      }
      sM[wave][row] = mnew;
      sL[wave][row] = sL[wave][row] * corr + sum;
      sC[wave][row] = corr;
    }

    float cr[8];
#pragma unroll
    for (int r = 0; r < 8; ++r) cr[r] = sC[wave][hi + r];
#pragma unroll
    for (int nt = 0; nt < 8; ++nt)
#pragma unroll
      for (int r = 0; r < 8; ++r) o[nt][r] *= cr[r];

    // O += P * V
    v16bf pf = load_a_frag(&sP[wave][0][0], 40);
#pragma unroll
    for (int nt = 0; nt < 8; ++nt) {
      v16bf vf = load_bT_frag(&pV[nt * 16][0], 40);
      o[nt] = wmma_bf16(pf, vf, o[nt]);
    }

#if USE_TDM
    if (wave == 0) __builtin_amdgcn_s_wait_tensorcnt(0);
    __syncthreads();
#endif
  }

  float inv[8];
#pragma unroll
  for (int r = 0; r < 8; ++r) inv[r] = 1.0f / sL[wave][hi + r];
  const int n = lane & 15;
  __bf16* Op = Oc + ((size_t)b * Sc + q0) * (Hc * HDc) + h * HDc;
#pragma unroll
  for (int nt = 0; nt < 8; ++nt)
#pragma unroll
    for (int r = 0; r < 8; ++r)
      Op[(size_t)(hi + r) * (Hc * HDc) + nt * 16 + n] =
          (__bf16)(o[nt][r] * inv[r]);
}

// ---------------------------------------------------------------------------
// Host launcher
// ---------------------------------------------------------------------------
extern "C" void kernel_launch(void* const* d_in, const int* in_sizes, int n_in,
                              void* d_out, int out_size, void* d_ws,
                              size_t ws_size, hipStream_t stream) {
  const float* x    = (const float*)d_in[0];
  const float* cosb = (const float*)d_in[1];
  const float* sinb = (const float*)d_in[2];
  const float* Wq   = (const float*)d_in[3];
  const float* Wk   = (const float*)d_in[4];
  const float* Wv   = (const float*)d_in[5];
  const float* Wo   = (const float*)d_in[6];
  float* out = (float*)d_out;

  const size_t nX  = (size_t)Bc * Sc * Dc;
  const size_t nWq = (size_t)Dc * Hc * HDc;
  const size_t nWk = (size_t)Dc * KVc * HDc;
  const size_t nWo = (size_t)Hc * HDc * Dc;
  const size_t nQ  = (size_t)Bc * Sc * Hc * HDc;
  const size_t nK  = (size_t)Bc * Sc * KVc * HDc;

  char*  ws  = (char*)d_ws;
  size_t off = 0;
  auto alloc = [&](size_t elems) {
    __bf16* p = (__bf16*)(ws + off);
    off += elems * sizeof(__bf16);
    return p;
  };
  __bf16* xb  = alloc(nX);
  __bf16* wqt = alloc(nWq);
  __bf16* wkt = alloc(nWk);
  __bf16* wvt = alloc(nWk);
  __bf16* wot = alloc(nWo);
  __bf16* qb  = alloc(nQ);
  __bf16* kb  = alloc(nK);
  __bf16* vb  = alloc(nK);
  __bf16* vtb = alloc(nK);
  __bf16* ab  = alloc(nQ);

  cvt_f32_bf16<<<(unsigned)(nX / 256), 256, 0, stream>>>(x, xb, (long)nX);
  transpose_cvt_f32_bf16<<<dim3(64, 64), 256, 0, stream>>>(Wq, wqt, Dc, Hc * HDc);
  transpose_cvt_f32_bf16<<<dim3(32, 64), 256, 0, stream>>>(Wk, wkt, Dc, KVc * HDc);
  transpose_cvt_f32_bf16<<<dim3(32, 64), 256, 0, stream>>>(Wv, wvt, Dc, KVc * HDc);
  transpose_cvt_f32_bf16<<<dim3(64, 64), 256, 0, stream>>>(Wo, wot, Hc * HDc, Dc);

  gemm_bf16<2048, 2048, true ><<<dim3(16, 32), 256, 0, stream>>>(xb, wqt, qb);
  gemm_bf16<1024, 2048, true ><<<dim3(8,  32), 256, 0, stream>>>(xb, wkt, kb);
  gemm_bf16<1024, 2048, true ><<<dim3(8,  32), 256, 0, stream>>>(xb, wvt, vb);

  rope_kernel<<<(unsigned)((size_t)Bc * Sc * Hc  * 64 / 256), 256, 0, stream>>>(qb, cosb, sinb, Hc);
  rope_kernel<<<(unsigned)((size_t)Bc * Sc * KVc * 64 / 256), 256, 0, stream>>>(kb, cosb, sinb, KVc);

  transpose_v_bf16<<<dim3(32, 64, 2), 256, 0, stream>>>(vb, vtb);

  attn_kernel<<<dim3(Sc / 64, Bc * Hc), 128, 0, stream>>>(qb, kb, vtb, ab);

  gemm_bf16<2048, 2048, false><<<dim3(16, 32), 256, 0, stream>>>(ab, wot, out);
}
